// MultiHeadAttention_58540404244505
// MI455X (gfx1250) — matrix operand order repaired and checked by reading, not compiled, not run on hardware
//
#include <hip/hip_runtime.h>

// Problem dims (fixed by the reference)
#define B_  2
#define S_  2048
#define E_  1024
#define H_  16
#define DH_ 64   // E/H

#if defined(__has_builtin)
#if __has_builtin(__builtin_amdgcn_global_load_async_to_lds_b128)
#define HAVE_ASYNC_LDS 1
#endif
#endif
#ifndef HAVE_ASYNC_LDS
#define HAVE_ASYNC_LDS 0
#endif

typedef __bf16 bf16_t;
typedef __attribute__((ext_vector_type(16))) __bf16 v16bf;
typedef __attribute__((ext_vector_type(8)))  __bf16 v8bf;
typedef __attribute__((ext_vector_type(8)))  float  v8f;
typedef int i32x4_ty __attribute__((vector_size(16)));

// One 16-byte global->LDS copy; async (ASYNCcnt-tracked) when available.
static __device__ __forceinline__ void async_copy16(const bf16_t* g, bf16_t* l) {
#if HAVE_ASYNC_LDS
  __builtin_amdgcn_global_load_async_to_lds_b128(
      (__attribute__((address_space(1))) i32x4_ty*)(bf16_t*)g,
      (__attribute__((address_space(3))) i32x4_ty*)l, 0, 0);
#else
  *(uint4*)l = *(const uint4*)g;
#endif
}

static __device__ __forceinline__ v8f wmma_bf16(v16bf a, v16bf b, v8f c) {
  // D = A(16x32 bf16) * B(32x16 bf16) + C(16x16 f32)
  return __builtin_amdgcn_wmma_f32_16x16x32_bf16(false, a, false, b, (short)0, c, false, false);
}

static __device__ __forceinline__ v16bf cat8(v8bf lo, v8bf hi) {
  return __builtin_shufflevector(lo, hi, 0, 1, 2, 3, 4, 5, 6, 7, 8, 9, 10, 11,
                                 12, 13, 14, 15);
}

// 16 contiguous bf16 at p (16B aligned) -> fragment (two b128 loads).
static __device__ __forceinline__ v16bf load_frag16(const bf16_t* p) {
  const v8bf* q = (const v8bf*)p;
  return cat8(q[0], q[1]);
}

// A-fragment lane data: 8 bf16 at p, 8 bf16 at p+16 (two b128 loads).
static __device__ __forceinline__ v16bf load_fragA(const bf16_t* p) {
  return cat8(*(const v8bf*)p, *(const v8bf*)(p + 16));
}

// ---- Fragment loaders (wave32 layouts per CDNA5 ISA 7.12.2) -----------------
// A fragment 16x32 (MxK): lane L<16 -> row M=L, K in {0..7, 16..23};
//                         lane L>=16 -> row M=L-16, K in {8..15, 24..31}.
static __device__ __forceinline__ v16bf load_A_bf16(const bf16_t* __restrict__ base,
                                                    int ld, int lane) {
  const int m  = lane & 15;
  const int kh = ((lane >> 4) & 1) * 8;
  return load_fragA(base + (size_t)m * ld + kh);
}

static __device__ __forceinline__ v16bf load_A_f32(const float* __restrict__ base,
                                                   int ld, int lane) {
  const int m  = lane & 15;
  const int kh = ((lane >> 4) & 1) * 8;
  const float* p = base + (size_t)m * ld + kh;   // 16B aligned
  const float4 f0 = ((const float4*)p)[0];
  const float4 f1 = ((const float4*)p)[1];
  const float4 f2 = ((const float4*)(p + 16))[0];
  const float4 f3 = ((const float4*)(p + 16))[1];
  v16bf a;
  a[0]  = (__bf16)f0.x; a[1]  = (__bf16)f0.y; a[2]  = (__bf16)f0.z; a[3]  = (__bf16)f0.w;
  a[4]  = (__bf16)f1.x; a[5]  = (__bf16)f1.y; a[6]  = (__bf16)f1.z; a[7]  = (__bf16)f1.w;
  a[8]  = (__bf16)f2.x; a[9]  = (__bf16)f2.y; a[10] = (__bf16)f2.z; a[11] = (__bf16)f2.w;
  a[12] = (__bf16)f3.x; a[13] = (__bf16)f3.y; a[14] = (__bf16)f3.z; a[15] = (__bf16)f3.w;
  return a;
}

// B fragment 32x16 (KxN): lane L<16 -> col N=L, K in {0..7, 16..23}; lane L>=16 ->
// col N=L-16, K in {8..15, 24..31}. Source stored "n-major": element (k,n) at
// base[n*ldn + k] -> each lane reads 8 bf16 at p and 8 at p+16.
static __device__ __forceinline__ v16bf load_B_nmajor(const bf16_t* __restrict__ base,
                                                      int ldn, int lane) {
  const int n  = lane & 15;
  const int kh = ((lane >> 4) & 1) * 8;
  return load_fragA(base + (size_t)n * ldn + kh);
}

// Reductions over a 16-lane group (rows m / m+8 live in lanes 0-15 / 16-31).
static __device__ __forceinline__ float group16_max(float v) {
#pragma unroll
  for (int m = 1; m <= 8; m <<= 1) v = fmaxf(v, __shfl_xor(v, m, 32));
  return v;
}
static __device__ __forceinline__ float group16_sum(float v) {
#pragma unroll
  for (int m = 1; m <= 8; m <<= 1) v += __shfl_xor(v, m, 32);
  return v;
}

// ---- Kernel 1: fused per-head QKV projection --------------------------------
// grid = (S/256, B*H, 3), block = 256 (8 waves, 32 rows per wave).
// Q out: [B*H, S, DH] pre-scaled by 1/sqrt(DH).  K out: [B*H, S, DH].
// V out: TRANSPOSED [B*H, DH, S] so attention can stage it with b128 copies.
__global__ __launch_bounds__(256) void qkv_proj_kernel(
    const float* __restrict__ q, const float* __restrict__ k,
    const float* __restrict__ v,
    const float* __restrict__ Wq, const float* __restrict__ bq,
    const float* __restrict__ Wk, const float* __restrict__ bk,
    const float* __restrict__ Wv, const float* __restrict__ bv,
    bf16_t* __restrict__ Qh, bf16_t* __restrict__ Kh, bf16_t* __restrict__ VhT) {
  __shared__ bf16_t wt[2][64 * 40];  // double-buffered W slab (n-major, padded)

  const int lane = threadIdx.x & 31;
  const int wave = threadIdx.x >> 5;
  const int bh   = blockIdx.y;
  const int b    = bh / H_;
  const int h    = bh % H_;

  const float* x; const float* W; const float* bias;
  if (blockIdx.z == 0)      { x = q; W = Wq; bias = bq; }
  else if (blockIdx.z == 1) { x = k; W = Wk; bias = bk; }
  else                      { x = v; W = Wv; bias = bv; }

  const int row0 = blockIdx.x * 256 + wave * 32;   // 2 strips of 16 rows
  const float* xrow = x + (size_t)b * S_ * E_ + (size_t)row0 * E_;
  const float* Wh   = W + (size_t)h * E_ * DH_;   // [E][DH] row-major

  // Stage one 32xDH W slab, transposed to n-major, 2 float4 per thread.
  auto stageW = [&](int bufi, int kk) {
    const float* ws = Wh + (size_t)kk * DH_;      // contiguous 32*64 fp32
#pragma unroll
    for (int j = 0; j < 2; ++j) {
      const int f  = ((int)threadIdx.x + 256 * j) * 4;
      const int kq = f >> 6, nq = f & 63;
      const float4 w4 = *(const float4*)(ws + f);
      wt[bufi][(nq + 0) * 40 + kq] = (__bf16)w4.x;
      wt[bufi][(nq + 1) * 40 + kq] = (__bf16)w4.y;
      wt[bufi][(nq + 2) * 40 + kq] = (__bf16)w4.z;
      wt[bufi][(nq + 3) * 40 + kq] = (__bf16)w4.w;
    }
  };

  v8f acc[2][4];
#pragma unroll
  for (int s = 0; s < 2; ++s)
#pragma unroll
    for (int t = 0; t < 4; ++t) acc[s][t] = (v8f){};

  stageW(0, 0);
  int buf = 0;
  for (int kk = 0; kk < E_; kk += 32, buf ^= 1) {
    __syncthreads();
    if (kk + 32 < E_) {
      stageW(buf ^ 1, kk + 32);
      __builtin_prefetch(Wh + (size_t)(kk + 64) * DH_, 0, 0);
    }
    const v16bf a0 = load_A_f32(xrow + kk, E_, lane);
    const v16bf a1 = load_A_f32(xrow + (size_t)16 * E_ + kk, E_, lane);
#pragma unroll
    for (int t = 0; t < 4; ++t) {
      const v16bf bfrag = load_B_nmajor(&wt[buf][t * 16 * 40], 40, lane);
      acc[0][t] = wmma_bf16(a0, bfrag, acc[0][t]);
      acc[1][t] = wmma_bf16(a1, bfrag, acc[1][t]);
    }
  }

  // Epilogue.
  const int n     = lane & 15;
  const int half8 = ((lane >> 4) & 1) * 8;
#pragma unroll
  for (int s = 0; s < 2; ++s) {
    const int rs = row0 + s * 16;
    if (blockIdx.z == 2) {
      // V: transposed store [bh][dh][s]; lane's 8 rows are consecutive s ->
      // one b128 store per tile.
      bf16_t* obase = VhT + (size_t)bh * DH_ * S_;
#pragma unroll
      for (int t = 0; t < 4; ++t) {
        const float bb = bias[h * DH_ + t * 16 + n];
        v8bf pk;
#pragma unroll
        for (int i = 0; i < 8; ++i) pk[i] = (__bf16)(acc[s][t][i] + bb);
        *(v8bf*)&obase[(size_t)(t * 16 + n) * S_ + rs + half8] = pk;
      }
    } else {
      const float osc = (blockIdx.z == 0) ? 0.125f : 1.0f;  // fold 1/sqrt(DH)
      bf16_t* out = (blockIdx.z == 0) ? Qh : Kh;
      bf16_t* orow = out + ((size_t)bh * S_ + rs) * DH_;
#pragma unroll
      for (int t = 0; t < 4; ++t) {
        const float bb = bias[h * DH_ + t * 16 + n];
#pragma unroll
        for (int i = 0; i < 8; ++i) {
          orow[(size_t)(i + half8) * DH_ + t * 16 + n] =
              (__bf16)((acc[s][t][i] + bb) * osc);
        }
      }
    }
  }
}

// ---- Kernel 2: flash attention ----------------------------------------------
// grid = (S/128, B*H), block = 128 (4 waves; each wave owns 32 query rows,
// two 16-row strips). Online softmax; K/V tiles double-buffered; staged via
// async-to-LDS (ASYNCcnt) when the toolchain exposes the builtin.
__global__ __launch_bounds__(128) void flash_attn_kernel(
    const bf16_t* __restrict__ Qh, const bf16_t* __restrict__ Kh,
    const bf16_t* __restrict__ VhT, bf16_t* __restrict__ ctx) {
  __shared__ bf16_t Kt[2][64 * 72];    // [key][dh], padded ld=72
  __shared__ bf16_t Vt[2][64 * 72];    // [dh][key], padded ld=72
  __shared__ bf16_t Pbuf[4][32 * 72];  // per-wave P staging (A layout source)

  const int lane  = threadIdx.x & 31;
  const int wave  = threadIdx.x >> 5;
  const int bh    = blockIdx.y;
  const int q0    = blockIdx.x * 128;
  const int n     = lane & 15;
  const int half8 = ((lane >> 4) & 1) * 8;

  const bf16_t* Ksrc = Kh + (size_t)bh * S_ * DH_;    // [key][dh]
  const bf16_t* Vsrc = VhT + (size_t)bh * DH_ * S_;   // [dh][s]

  // Stage one 64-key K tile and V tile: 4 x 16B each per thread.
  auto stageKV = [&](int bufi, int kb) {
#pragma unroll
    for (int j = 0; j < 4; ++j) {
      const int idx = (int)threadIdx.x + 128 * j;  // 512 chunks of 16B
      const int row = idx >> 3;
      const int c   = (idx & 7) * 8;
      async_copy16(Ksrc + ((size_t)(kb + row)) * DH_ + c,
                   &Kt[bufi][row * 72 + c]);
      async_copy16(Vsrc + (size_t)row * S_ + kb + c,
                   &Vt[bufi][row * 72 + c]);
    }
  };

  // Q fragments straight from global (Q pre-scaled by 1/sqrt(DH)).
  const bf16_t* Qsrc = Qh + ((size_t)bh * S_ + q0 + wave * 32) * DH_;
  v16bf a_q[2][2];  // [strip][kstep]
#pragma unroll
  for (int st = 0; st < 2; ++st)
#pragma unroll
    for (int s = 0; s < 2; ++s)
      a_q[st][s] =
          load_A_bf16(Qsrc + (size_t)st * 16 * DH_ + s * 32, DH_, lane);

  float rowmax[2][8], rowsum[2][8];
#pragma unroll
  for (int st = 0; st < 2; ++st)
#pragma unroll
    for (int i = 0; i < 8; ++i) { rowmax[st][i] = -1e30f; rowsum[st][i] = 0.0f; }
  v8f o[2][4];
#pragma unroll
  for (int st = 0; st < 2; ++st)
#pragma unroll
    for (int t = 0; t < 4; ++t) o[st][t] = (v8f){};

  stageKV(0, 0);
  int buf = 0;
  for (int kb = 0; kb < S_; kb += 64, buf ^= 1) {
#if HAVE_ASYNC_LDS
    asm volatile("s_wait_asynccnt 0" ::: "memory");
#endif
    __syncthreads();
    if (kb + 64 < S_) {
      stageKV(buf ^ 1, kb + 64);
      __builtin_prefetch(Ksrc + (size_t)(kb + 128) * DH_, 0, 0);
    }

#pragma unroll
    for (int st = 0; st < 2; ++st) {
      // S = Q * K^T. K^T element (k=dh, n=key) = Kt[n*72 + k] (n-major).
      v8f sc[4];
#pragma unroll
      for (int t = 0; t < 4; ++t) sc[t] = (v8f){};
#pragma unroll
      for (int t = 0; t < 4; ++t) {
#pragma unroll
        for (int ksi = 0; ksi < 2; ++ksi) {
          const v16bf bk =
              load_B_nmajor(&Kt[buf][t * 16 * 72 + ksi * 32], 72, lane);
          sc[t] = wmma_bf16(a_q[st][ksi], bk, sc[t]);
        }
      }

      // Online softmax update (row-wise across 16-lane groups).
#pragma unroll
      for (int i = 0; i < 8; ++i) {
        float v0 = fmaxf(fmaxf(sc[0][i], sc[1][i]), fmaxf(sc[2][i], sc[3][i]));
        const float mnew = fmaxf(rowmax[st][i], group16_max(v0));
        const float fac = __expf(rowmax[st][i] - mnew);
        rowmax[st][i] = mnew;
        float ps = 0.0f;
#pragma unroll
        for (int t = 0; t < 4; ++t) {
          const float e = __expf(sc[t][i] - mnew);
          sc[t][i] = e;
          ps += e;
        }
        rowsum[st][i] = rowsum[st][i] * fac + group16_sum(ps);
#pragma unroll
        for (int t = 0; t < 4; ++t) o[st][t][i] *= fac;
      }

      // P (C/D layout) -> LDS, strip-local region.
#pragma unroll
      for (int t = 0; t < 4; ++t)
#pragma unroll
        for (int i = 0; i < 8; ++i)
          Pbuf[wave][(st * 16 + i + half8) * 72 + t * 16 + n] =
              (__bf16)sc[t][i];
    }
    asm volatile("s_wait_dscnt 0" ::: "memory");

    // O += P * V.  V element (k=key, n=dh) = Vt[n*72 + k] (n-major).
#pragma unroll
    for (int t = 0; t < 4; ++t) {
#pragma unroll
      for (int ksi = 0; ksi < 2; ++ksi) {
        const v16bf bv =
            load_B_nmajor(&Vt[buf][t * 16 * 72 + ksi * 32], 72, lane);
#pragma unroll
        for (int st = 0; st < 2; ++st) {
          const v16bf ap =
              load_A_bf16(&Pbuf[wave][(st * 16) * 72 + ksi * 32], 72, lane);
          o[st][t] = wmma_bf16(ap, bv, o[st][t]);
        }
      }
    }
  }

  // Epilogue: normalize and write ctx[b, s, h*DH + d] (bf16).
  const int b = bh / H_;
  const int h = bh % H_;
#pragma unroll
  for (int st = 0; st < 2; ++st) {
    bf16_t* crow = ctx + ((size_t)b * S_ + q0 + wave * 32 + st * 16) * E_ +
                   (size_t)h * DH_;
    float inv[8];
#pragma unroll
    for (int i = 0; i < 8; ++i) inv[i] = 1.0f / rowsum[st][i];
#pragma unroll
    for (int t = 0; t < 4; ++t)
#pragma unroll
      for (int i = 0; i < 8; ++i)
        crow[(size_t)(i + half8) * E_ + t * 16 + n] =
            (__bf16)(o[st][t][i] * inv[i]);
  }
}

// ---- Kernel 3: output projection  out = ctx @ Wo + bo -----------------------
// grid = (B*S/256, E/64), block = 256 (8 waves, 32 rows per wave).
__global__ __launch_bounds__(256) void out_proj_kernel(
    const bf16_t* __restrict__ ctx, const float* __restrict__ Wo,
    const float* __restrict__ bo, float* __restrict__ out) {
  __shared__ bf16_t wt[2][64 * 40];  // double-buffered Wo slab (n-major)

  const int lane = threadIdx.x & 31;
  const int wave = threadIdx.x >> 5;
  const int row0 = blockIdx.x * 256 + wave * 32;
  const int col0 = blockIdx.y * 64;

  auto stageW = [&](int bufi, int kk) {
#pragma unroll
    for (int j = 0; j < 2; ++j) {
      const int f = ((int)threadIdx.x + 256 * j) * 4;
      const int r = f >> 6, c = f & 63;  // (k, n..n+3)
      const float4 w4 = *(const float4*)(Wo + (size_t)(kk + r) * E_ + col0 + c);
      wt[bufi][(c + 0) * 40 + r] = (__bf16)w4.x;
      wt[bufi][(c + 1) * 40 + r] = (__bf16)w4.y;
      wt[bufi][(c + 2) * 40 + r] = (__bf16)w4.z;
      wt[bufi][(c + 3) * 40 + r] = (__bf16)w4.w;
    }
  };

  v8f acc[2][4];
#pragma unroll
  for (int s = 0; s < 2; ++s)
#pragma unroll
    for (int t = 0; t < 4; ++t) acc[s][t] = (v8f){};

  stageW(0, 0);
  int buf = 0;
  for (int kk = 0; kk < E_; kk += 32, buf ^= 1) {
    __syncthreads();
    if (kk + 32 < E_) {
      stageW(buf ^ 1, kk + 32);
      __builtin_prefetch(Wo + (size_t)(kk + 64) * E_ + col0, 0, 0);
    }
    const v16bf a0 = load_A_bf16(ctx + (size_t)row0 * E_ + kk, E_, lane);
    const v16bf a1 =
        load_A_bf16(ctx + (size_t)(row0 + 16) * E_ + kk, E_, lane);
#pragma unroll
    for (int t = 0; t < 4; ++t) {
      const v16bf bfrag = load_B_nmajor(&wt[buf][t * 16 * 40], 40, lane);
      acc[0][t] = wmma_bf16(a0, bfrag, acc[0][t]);
      acc[1][t] = wmma_bf16(a1, bfrag, acc[1][t]);
    }
  }

  const int n     = lane & 15;
  const int half8 = ((lane >> 4) & 1) * 8;
#pragma unroll
  for (int s = 0; s < 2; ++s) {
#pragma unroll
    for (int t = 0; t < 4; ++t) {
      const float bb = bo[col0 + t * 16 + n];
#pragma unroll
      for (int i = 0; i < 8; ++i) {
        out[(size_t)(row0 + s * 16 + i + half8) * E_ + col0 + t * 16 + n] =
            acc[s][t][i] + bb;
      }
    }
  }
}

// ---- Host-side launch -------------------------------------------------------
extern "C" void kernel_launch(void* const* d_in, const int* in_sizes, int n_in,
                              void* d_out, int out_size, void* d_ws,
                              size_t ws_size, hipStream_t stream) {
  (void)in_sizes; (void)n_in; (void)out_size; (void)ws_size;

  const float* q  = (const float*)d_in[0];
  const float* k  = (const float*)d_in[1];
  const float* v  = (const float*)d_in[2];
  const float* Wq = (const float*)d_in[3];
  const float* bq = (const float*)d_in[4];
  const float* Wk = (const float*)d_in[5];
  const float* bk = (const float*)d_in[6];
  const float* Wv = (const float*)d_in[7];
  const float* bv = (const float*)d_in[8];
  const float* Wo = (const float*)d_in[9];
  const float* bo = (const float*)d_in[10];
  float* out = (float*)d_out;

  // Workspace: bf16 Qh/Kh [B*H,S,DH], VhT [B*H,DH,S], ctx [B,S,E] => 32 MB.
  const size_t qkv_elems = (size_t)B_ * H_ * S_ * DH_;  // == B*S*E
  bf16_t* Qh  = (bf16_t*)d_ws;
  bf16_t* Kh  = Qh + qkv_elems;
  bf16_t* VhT = Kh + qkv_elems;
  bf16_t* ctx = VhT + qkv_elems;

  qkv_proj_kernel<<<dim3(S_ / 256, B_ * H_, 3), 256, 0, stream>>>(
      q, k, v, Wq, bq, Wk, bk, Wv, bv, Qh, Kh, VhT);

  flash_attn_kernel<<<dim3(S_ / 128, B_ * H_), 128, 0, stream>>>(Qh, Kh, VhT,
                                                                 ctx);

  out_proj_kernel<<<dim3((B_ * S_) / 256, E_ / 64), 256, 0, stream>>>(ctx, Wo,
                                                                      bo, out);
}
